// IMSE_20074677141643
// MI455X (gfx1250) — matrix operand order repaired and checked by reading, not compiled, not run on hardware
//
#include <hip/hip_runtime.h>
#include <hip/hip_bf16.h>

typedef _Float16 v16h __attribute__((ext_vector_type(16)));
typedef _Float16 h8   __attribute__((ext_vector_type(8)));
typedef float    v8f  __attribute__((ext_vector_type(8)));

#define NEG_INF (-3.402823466e38f)

// Problem geometry (from reference)
constexpr int QM    = 75;          // queries
constexpr int QFS   = 225;         // 15*15 query positions
constexpr int CDIM  = 640;         // channels (K)
constexpr int NWAY  = 5;
constexpr int PM    = 5;           // supports per class
constexpr int PFS   = 225;         // 15*15 support positions
constexpr int PPAD  = 256;         // padded to 16 N-tiles (branch-free waves)
constexpr int MROWS = QM * QFS;    // 16875
constexpr int MPAD  = 16896;       // multiple of 32 (block M-tile)
constexpr int MBLKS = MPAD / 32;   // 528
constexpr int NTILES = PPAD / 16;  // 16
constexpr int CS     = NWAY * PM;  // 25 (class*support combos)

// Workspace layout (bytes, 256-aligned)
constexpr size_t OFF_QN = 0;                                     // MPAD*CDIM halves
constexpr size_t QN_BYTES = (size_t)MPAD * CDIM * 2;             // 21,626,880
constexpr size_t OFF_PN = OFF_QN + QN_BYTES;                     // CS*PPAD*CDIM halves
constexpr size_t PN_BYTES = (size_t)CS * PPAD * CDIM * 2;        // 8,192,000
constexpr size_t OFF_TV = OFF_PN + PN_BYTES;                     // CS*MROWS*3 floats
constexpr size_t TV_BYTES = ((size_t)CS * MROWS * 3 * 4 + 255) & ~(size_t)255;
constexpr size_t OFF_TI = OFF_TV + TV_BYTES;                     // CS*MROWS*3 ints

__global__ void zero_out_kernel(float* __restrict__ out, int n) {
    int i = blockIdx.x * blockDim.x + threadIdx.x;
    if (i < n) out[i] = 0.0f;
}

// Normalize query descriptors: x1 [QM][C][15][15] -> qn [MPAD][C] f16, rows >= MROWS zeroed
__global__ __launch_bounds__(128)
void norm_q_kernel(const float* __restrict__ x1, _Float16* __restrict__ qn) {
    __shared__ float red[128];
    const int r = blockIdx.x;
    const int t = threadIdx.x;
    _Float16* dst = qn + (size_t)r * CDIM;
    if (r >= MROWS) {                      // zero pad rows (uniform branch per block)
        for (int e = t; e < CDIM; e += 128) dst[e] = (_Float16)0.0f;
        return;
    }
    const int i  = r / QFS, j = r % QFS;
    const int jh = j / 15,  jw = j % 15;
    const float* src = x1 + (size_t)i * CDIM * 225 + jh * 15 + jw;  // stride 225 per channel
    float vals[5]; float ss = 0.0f;
    #pragma unroll
    for (int q = 0; q < 5; ++q) {
        float v = src[(size_t)(t + 128 * q) * 225];
        vals[q] = v; ss += v * v;
    }
    red[t] = ss; __syncthreads();
    for (int off = 64; off > 0; off >>= 1) {
        if (t < off) red[t] += red[t + off];
        __syncthreads();
    }
    const float rn = rsqrtf(red[0]);
    #pragma unroll
    for (int q = 0; q < 5; ++q) dst[t + 128 * q] = (_Float16)(vals[q] * rn);
}

// Normalize support descriptors: x2 [NWAY][PM][C][15][15] -> pn [CS*PPAD][C] f16, p>=PFS zeroed
__global__ __launch_bounds__(128)
void norm_p_kernel(const float* __restrict__ x2, _Float16* __restrict__ pn) {
    __shared__ float red[128];
    const int b = blockIdx.x;                 // b = (cls*PM + s)*PPAD + p
    const int t = threadIdx.x;
    const int p = b % PPAD;
    const int cs = b / PPAD;                  // cls*PM + s
    _Float16* dst = pn + (size_t)b * CDIM;
    if (p >= PFS) {
        for (int e = t; e < CDIM; e += 128) dst[e] = (_Float16)0.0f;
        return;
    }
    const int ph = p / 15, pw = p % 15;
    const float* src = x2 + (size_t)cs * CDIM * 225 + ph * 15 + pw;
    float vals[5]; float ss = 0.0f;
    #pragma unroll
    for (int q = 0; q < 5; ++q) {
        float v = src[(size_t)(t + 128 * q) * 225];
        vals[q] = v; ss += v * v;
    }
    red[t] = ss; __syncthreads();
    for (int off = 64; off > 0; off >>= 1) {
        if (t < off) red[t] += red[t + off];
        __syncthreads();
    }
    const float rn = rsqrtf(red[0]);
    #pragma unroll
    for (int q = 0; q < 5; ++q) dst[t + 128 * q] = (_Float16)(vals[q] * rn);
}

// Fused GEMM (WMMA f16->f32) + top-3 over support positions.
// Block = 256 threads (8 waves, 2x4): block tile = 32 query rows x 256 support cols
// for one (class, support) pair. Each wave: 1 A-load feeds 4 WMMAs per K-step.
// sim never touches HBM; top-3 reduction runs in the same block's LDS.
__global__ __launch_bounds__(256)
void gemm_topk_kernel(const _Float16* __restrict__ qn, const _Float16* __restrict__ pn,
                      float* __restrict__ tv, int* __restrict__ ti) {
    __shared__ float sim[32][PPAD];          // 32,768 B
    const int tid  = threadIdx.x;
    const int wave = tid >> 5;
    const int lane = tid & 31;
    const int half = lane >> 4;
    const int l15  = lane & 15;
    const int mw   = wave >> 2;              // 0..1  (M-tile within block)
    const int nw   = wave & 3;               // 0..3  (group of 4 N-tiles)
    const int mbase = blockIdx.x * 32;
    const int cs    = blockIdx.y;            // cls*PM + s

    const _Float16* arow = qn + (size_t)(mbase + mw * 16 + l15) * CDIM;
    const _Float16* brow[4];
    #pragma unroll
    for (int u = 0; u < 4; ++u)
        brow[u] = pn + ((size_t)cs * PPAD + (nw * 4 + u) * 16 + l15) * CDIM;

    v8f acc[4] = {v8f{}, v8f{}, v8f{}, v8f{}};
    for (int kb = 0; kb < CDIM; kb += 32) {
        // A layout (16-bit A 16x32): lane m=l15; half 0 -> K 0..15; half 1 -> K 16..31
        h8 a0 = *(const h8*)(arow + kb + 16 * half);
        h8 a1 = *(const h8*)(arow + kb + 16 * half + 8);
        v16h a = __builtin_shufflevector(a0, a1, 0,1,2,3,4,5,6,7,8,9,10,11,12,13,14,15);
        #pragma unroll
        for (int u = 0; u < 4; ++u) {
            // B layout (16-bit B 32x16, same k order as A): lane n=l15; half 0 -> K 0..15; half 1 -> K 16..31
            v16h b = *(const v16h*)(brow[u] + kb + 16 * half);
            acc[u] = __builtin_amdgcn_wmma_f32_16x16x32_f16(false, a, false, b,
                                                            (short)0, acc[u], false, false);
        }
    }
    // C/D layout: element r of acc in lane -> M = r + 8*half, N = l15
    #pragma unroll
    for (int u = 0; u < 4; ++u)
        #pragma unroll
        for (int r = 0; r < 8; ++r)
            sim[mw * 16 + r + 8 * half][(nw * 4 + u) * 16 + l15] = acc[u][r];
    __syncthreads();

    // Parallel top-3 per row: 8 threads/row, each scans 32 strided elements.
    const int row = tid >> 3;                // 0..31
    const int c   = tid & 7;                 // 0..7
    float v0 = NEG_INF, v1 = NEG_INF, v2 = NEG_INF;
    int   i0 = 0, i1 = 0, i2 = 0;
    for (int q = 0; q < 32; ++q) {
        const int p = c + 8 * q;
        const float v = (p < PFS) ? sim[row][p] : NEG_INF;  // mask pad columns
        if (v > v0)      { v2 = v1; i2 = i1; v1 = v0; i1 = i0; v0 = v; i0 = p; }
        else if (v > v1) { v2 = v1; i2 = i1; v1 = v;  i1 = p; }
        else if (v > v2) { v2 = v;  i2 = p; }
    }
    __syncthreads();                    // sim no longer needed; repurpose LDS
    float* cf = &sim[0][0];
    int*   ci = (int*)(cf + 768);
    const int cbase = (row * 8 + c) * 3;
    cf[cbase + 0] = v0; cf[cbase + 1] = v1; cf[cbase + 2] = v2;
    ci[cbase + 0] = i0; ci[cbase + 1] = i1; ci[cbase + 2] = i2;
    __syncthreads();
    if (c == 0) {                       // 32 leaders merge 24 candidates each
        const int m_global = mbase + row;
        if (m_global < MROWS) {
            const size_t obase = ((size_t)cs * MROWS + m_global) * 3;
            #pragma unroll
            for (int k = 0; k < 3; ++k) {
                float best = NEG_INF; int bj = 0;
                for (int cc = 0; cc < 24; ++cc) {
                    const float v = cf[row * 24 + cc];
                    if (v > best) { best = v; bj = cc; }
                }
                tv[obase + k] = best;
                ti[obase + k] = ci[row * 24 + bj];
                cf[row * 24 + bj] = NEG_INF;
            }
        }
    }
}

// Scatter sparse values into [p][15x15] slices, run (3x3 gauss conv -> 2x2 maxpool) x2,
// sum, and accumulate into out[i*NWAY + cls]. One block per (i, cls, s).
__global__ __launch_bounds__(256)
void score_kernel(const float* __restrict__ tv, const int* __restrict__ ti,
                  float* __restrict__ out) {
    __shared__ float chunk[64][225];    // 57,600 B
    __shared__ float red[256];
    const int b = blockIdx.x;           // 0 .. QM*CS-1
    const int i   = b / CS;
    const int rem = b % CS;             // cls*PM + s
    const int cls = rem / PM;
    const int t = threadIdx.x;
    const size_t tb = ((size_t)rem * MROWS + (size_t)i * QFS) * 3;  // triples for j=0..224

    // 3x3 gaussian (KERNEL_SIZE=1, SIGMA=0.8), normalized to sum 1
    const float e1 = __expf(-0.625f);
    const float e2 = __expf(-1.25f);
    const float den = 1.0f + 4.0f * e1 + 4.0f * e2;
    const float gc = 1.0f / den, ge = e1 / den, gd = e2 / den;
    const float g[3][3] = {{gd, ge, gd}, {ge, gc, ge}, {gd, ge, gd}};

    float lsum = 0.0f;
    for (int cb = 0; cb < PFS; cb += 64) {
        const int ns = (PFS - cb < 64) ? (PFS - cb) : 64;
        for (int e = t; e < 64 * 225; e += 256) ((float*)chunk)[e] = 0.0f;
        __syncthreads();
        for (int e = t; e < QFS * 3; e += 256) {
            const int j = e / 3;
            const int p = ti[tb + e];
            if (p >= cb && p < cb + ns) chunk[p - cb][j] = tv[tb + e];
        }
        __syncthreads();
        if (t < ns) {
            const float* S = &chunk[t][0];      // 15x15 map over query positions
            float p1[6][6];
            for (int y1 = 0; y1 < 6; ++y1) {
                for (int x1 = 0; x1 < 6; ++x1) {
                    float mx = NEG_INF;
                    for (int sy = 0; sy < 2; ++sy)
                        for (int sx = 0; sx < 2; ++sx) {
                            const int cy = 2 * y1 + sy, cx = 2 * x1 + sx; // conv1: 13x13
                            float a = 0.0f;
                            #pragma unroll
                            for (int dy = 0; dy < 3; ++dy)
                                #pragma unroll
                                for (int dx = 0; dx < 3; ++dx)
                                    a += g[dy][dx] * S[(cy + dy) * 15 + (cx + dx)];
                            mx = fmaxf(mx, a);
                        }
                    p1[y1][x1] = mx;            // after pool1: 6x6
                }
            }
            for (int yy = 0; yy < 2; ++yy) {
                for (int xx = 0; xx < 2; ++xx) {
                    float mx = NEG_INF;
                    for (int sy = 0; sy < 2; ++sy)
                        for (int sx = 0; sx < 2; ++sx) {
                            const int cy = 2 * yy + sy, cx = 2 * xx + sx; // conv2: 4x4
                            float a = 0.0f;
                            #pragma unroll
                            for (int dy = 0; dy < 3; ++dy)
                                #pragma unroll
                                for (int dx = 0; dx < 3; ++dx)
                                    a += g[dy][dx] * p1[cy + dy][cx + dx];
                            mx = fmaxf(mx, a);
                        }
                    lsum += mx;                 // after pool2: 2x2, summed
                }
            }
        }
        __syncthreads();
    }
    red[t] = lsum; __syncthreads();
    for (int off = 128; off > 0; off >>= 1) {
        if (t < off) red[t] += red[t + off];
        __syncthreads();
    }
    if (t == 0) atomicAdd(&out[i * NWAY + cls], red[0]);
}

extern "C" void kernel_launch(void* const* d_in, const int* in_sizes, int n_in,
                              void* d_out, int out_size, void* d_ws, size_t ws_size,
                              hipStream_t stream) {
    const float* x1 = (const float*)d_in[0];   // [75,640,15,15]
    const float* x2 = (const float*)d_in[1];   // [5,5,640,15,15]
    float* out = (float*)d_out;                // [75,5]
    char* ws = (char*)d_ws;
    _Float16* qn = (_Float16*)(ws + OFF_QN);
    _Float16* pn = (_Float16*)(ws + OFF_PN);
    float*    tv = (float*)(ws + OFF_TV);
    int*      ti = (int*)(ws + OFF_TI);

    zero_out_kernel<<<(QM * NWAY + 255) / 256, 256, 0, stream>>>(out, QM * NWAY);
    norm_q_kernel<<<MPAD, 128, 0, stream>>>(x1, qn);
    norm_p_kernel<<<CS * PPAD, 128, 0, stream>>>(x2, pn);
    gemm_topk_kernel<<<dim3(MBLKS, CS), 256, 0, stream>>>(qn, pn, tv, ti);
    score_kernel<<<QM * CS, 256, 0, stream>>>(tv, ti, out);
}
